// EmbeddingSum_6932077216269
// MI455X (gfx1250) — matrix operand order repaired and checked by reading, not compiled, not run on hardware
//
#include <hip/hip_runtime.h>
#include <stdint.h>

// ---------------------------------------------------------------------------
// EmbeddingSum  ==  GEMM:  out[16384,64] = x[16384,10000] @ W[10000,64]
// x is {0,1} (exact in f16).  W split into f16 hi + f16 lo halves so
// f16-WMMA with f32 accumulation reproduces f32 precision.
// HBM-bound: streams 655 MB of x -> ~28 us floor @ 23.3 TB/s.
// Tiling for max concurrency: 16 rows/wave (N=64 kept per-wave so x is read
// exactly once) -> 1024 waves as 256 blocks x 128 threads.
// ---------------------------------------------------------------------------

typedef __attribute__((ext_vector_type(16))) _Float16 v16h;
typedef __attribute__((ext_vector_type(8)))  float    v8f;
typedef __attribute__((ext_vector_type(4)))  float    v4f;
typedef __attribute__((ext_vector_type(4)))  uint32_t v4u;

#define N_CODES 10000
#define EMBED   64
#define KTILES  313          // ceil(10000/32)
#define NTILES  4            // 64 / 16
#define ROWS    16384        // B*T
#define BLK_ROWS 64          // 4 waves x 16 rows
#define NBLOCKS  (ROWS / BLK_ROWS)   // 256

// ---------------------------------------------------------------------------
// Pack W (f32, row-major [K=10000, E=64]) into the CDNA5 WMMA B-operand
// layout for 32x16 f16 tiles, zero-padded to K=10016, hi + lo halves.
// Layout index: (((t*NTILES + nt)*32 + lane)*8 + j)  dwords,
//   lane 0-15  : K = t*32 + 2j (+8 if j>=4), +1      packed {lo,hi}
//   lane 16-31 : K = t*32 + 8 + 2j (+8 if j>=4), +1
//   column n   = nt*16 + (lane & 15)
// ---------------------------------------------------------------------------
__global__ void pack_W_kernel(const float* __restrict__ W,
                              uint32_t* __restrict__ Bhi,
                              uint32_t* __restrict__ Blo)
{
    const int total = KTILES * NTILES * 32 * 8;
    int idx = blockIdx.x * blockDim.x + threadIdx.x;
    if (idx >= total) return;

    int j    =  idx       & 7;
    int lane = (idx >> 3) & 31;
    int nt   = (idx >> 8) & 3;
    int t    =  idx >> 10;

    int n    = nt * 16 + (lane & 15);
    int krel = ((lane & 16) ? 8 : 0) + 2 * j + ((j & 4) ? 8 : 0);
    int k0   = t * 32 + krel;
    int k1   = k0 + 1;

    float w0 = (k0 < N_CODES) ? W[(size_t)k0 * EMBED + n] : 0.0f;
    float w1 = (k1 < N_CODES) ? W[(size_t)k1 * EMBED + n] : 0.0f;

    _Float16 h0 = (_Float16)w0;
    _Float16 h1 = (_Float16)w1;
    _Float16 l0 = (_Float16)(w0 - (float)h0);
    _Float16 l1 = (_Float16)(w1 - (float)h1);

    union { _Float16 h[2]; uint32_t u; } ph, pl;
    ph.h[0] = h0; ph.h[1] = h1;
    pl.h[0] = l0; pl.h[1] = l1;
    Bhi[idx] = ph.u;
    Blo[idx] = pl.u;
}

// Load one 32x16 f16 B fragment (8 dwords per lane) from packed layout.
__device__ inline v16h load_bfrag(const uint32_t* __restrict__ p)
{
    union { struct { v4u a, b; } u; v16h h; } c;
    c.u.a = *(const v4u*)(p);
    c.u.b = *(const v4u*)(p + 4);
    return c.h;
}

// Build A fragment per ISA 16-bit A 16x32 layout: elements 0..7 = K chunk at
// koff, elements 8..15 = K chunk at koff+16 (koff = 0 for lanes 0-15, 8 for
// lanes 16-31).  f32 -> f16 is exact because x is {0,1}.
__device__ inline v16h cvt_afrag(v4f a0, v4f a1, v4f a2, v4f a3)
{
    v16h r;
    r[0]  = (_Float16)a0[0]; r[1]  = (_Float16)a0[1];
    r[2]  = (_Float16)a0[2]; r[3]  = (_Float16)a0[3];
    r[4]  = (_Float16)a1[0]; r[5]  = (_Float16)a1[1];
    r[6]  = (_Float16)a1[2]; r[7]  = (_Float16)a1[3];
    r[8]  = (_Float16)a2[0]; r[9]  = (_Float16)a2[1];
    r[10] = (_Float16)a2[2]; r[11] = (_Float16)a2[3];
    r[12] = (_Float16)a3[0]; r[13] = (_Float16)a3[1];
    r[14] = (_Float16)a3[2]; r[15] = (_Float16)a3[3];
    return r;
}

__global__ __launch_bounds__(128)
void emb_wmma_kernel(const float*    __restrict__ x,
                     const uint32_t* __restrict__ Bhi,
                     const uint32_t* __restrict__ Blo,
                     float*          __restrict__ out)
{
    const int lane  = threadIdx.x & 31;
    const int wave  = threadIdx.x >> 5;                // 0..3
    const int mbase = blockIdx.x * BLK_ROWS + wave * 16; // 16 rows per wave
    const int arow  = mbase + (lane & 15);
    const int koff  = (lane >> 4) * 8;                 // 0 or 8
    const float* xr = x + (size_t)arow * N_CODES + koff;

    v8f acc[NTILES] = {};
    const v4f vzero = {0.0f, 0.0f, 0.0f, 0.0f};

    for (int t = 0; t < KTILES; ++t) {
        const float* p = xr + t * 32;

        // -------- A fragment (streams x from HBM) --------
        v4f a0 = *(const v4f*)(p);
        v4f a1 = *(const v4f*)(p + 4);
        v4f a2 = vzero, a3 = vzero;
        if (t < KTILES - 1) {                          // uniform: K tail = 16
            a2 = *(const v4f*)(p + 16);
            a3 = *(const v4f*)(p + 20);
            __builtin_prefetch(p + 32, 0, 1);          // global_prefetch_b8
        }
        v16h A = cvt_afrag(a0, a1, a2, a3);

        // -------- all 8 B fragments first (long load clause, L2-resident) ---
        const size_t bbase = ((size_t)(t * NTILES) * 32 + lane) * 8;
        v16h bh[NTILES], bl[NTILES];
#pragma unroll
        for (int nt = 0; nt < NTILES; ++nt) {
            const size_t o = bbase + (size_t)nt * 32 * 8;
            bh[nt] = load_bfrag(Bhi + o);
            bl[nt] = load_bfrag(Blo + o);
        }

        // -------- 8 WMMAs: hi + lo halves into the same f32 accumulator ----
#pragma unroll
        for (int nt = 0; nt < NTILES; ++nt) {
            acc[nt] = __builtin_amdgcn_wmma_f32_16x16x32_f16(
                false, A, false, bh[nt], (short)0, acc[nt], false, false);
            acc[nt] = __builtin_amdgcn_wmma_f32_16x16x32_f16(
                false, A, false, bl[nt], (short)0, acc[nt], false, false);
        }
    }

    // -------- store per C/D layout: VGPR i -> M = i (lanes 0-15) / i+8 ------
    const int ncol = lane & 15;
    const int rsel = (lane >> 4) * 8;
#pragma unroll
    for (int nt = 0; nt < NTILES; ++nt)
#pragma unroll
        for (int i = 0; i < 8; ++i)
            out[(size_t)(mbase + rsel + i) * EMBED + nt * 16 + ncol]
                = acc[nt][i];
}

extern "C" void kernel_launch(void* const* d_in, const int* in_sizes, int n_in,
                              void* d_out, int out_size, void* d_ws, size_t ws_size,
                              hipStream_t stream)
{
    const float* x = (const float*)d_in[0];   // (256,64,10000) f32
    const float* W = (const float*)d_in[1];   // (10000,64)     f32
    float* out = (float*)d_out;               // (256,64,64)    f32

    const size_t packed = (size_t)KTILES * NTILES * 32 * 8; // dwords per half
    uint32_t* Bhi = (uint32_t*)d_ws;
    uint32_t* Blo = Bhi + packed;

    const int total = (int)packed;
    pack_W_kernel<<<(total + 255) / 256, 256, 0, stream>>>(W, Bhi, Blo);
    emb_wmma_kernel<<<NBLOCKS, 128, 0, stream>>>(x, Bhi, Blo, out);
}
